// Encoder_22024592294223
// MI455X (gfx1250) — matrix operand order repaired and checked by reading, not compiled, not run on hardware
//
#include <hip/hip_runtime.h>
#include <hip/hip_bf16.h>

// ---------------------------------------------------------------------------
// ViT encoder forward for MI455X (gfx1250, wave32, WMMA, async LDS copies).
// L=12, D=768, DH=64, NH=12, P=16, B=16, img 384x384 -> grid 24x24, N=576.
// Rows of the "token matrix" = B*N = 9216.  All GEMM dims divide 64/64/32.
// ---------------------------------------------------------------------------

#define D_MODEL   768
#define N_TOK     576
#define B_BATCH   16
#define ROWS      (B_BATCH * N_TOK)   // 9216
#define NHEAD     12
#define DHEAD     64
#define FFN_DIM   3072
#define GW_GRID   24

typedef __attribute__((ext_vector_type(16))) __bf16 v16bf;
typedef __attribute__((ext_vector_type(8)))  float  v8f;

struct __align__(16) U4 { unsigned x, y, z, w; };
union Frag { U4 u[2]; v16bf v; };

__device__ __forceinline__ unsigned short f2bf(float f) {
    unsigned u = __float_as_uint(f);
    u += 0x7FFFu + ((u >> 16) & 1u);       // round-to-nearest-even
    return (unsigned short)(u >> 16);
}
__device__ __forceinline__ float bf2f(unsigned short h) {
    return __uint_as_float(((unsigned)h) << 16);
}

// Async global -> LDS copy, 16B per lane (CDNA5, tracked by ASYNCcnt).
// LDS destination address = low 32 bits of the generic shared pointer
// (generic LDS pointers are {shared_aperture_hi, lds_byte_offset}).
__device__ __forceinline__ void async_copy_b128(const void* gsrc, void* ldst) {
    unsigned lds32 = (unsigned)(unsigned long long)ldst;
    asm volatile("global_load_async_to_lds_b128 %0, %1, off"
                 :: "v"(lds32), "v"(gsrc) : "memory");
}
__device__ __forceinline__ void async_wait0() {
    asm volatile("s_wait_asynccnt 0x0" ::: "memory");
}

// -------------------------- LayerNorm helpers ------------------------------
__device__ __forceinline__ void block_ln_stats(float s, float q, float* red,
                                               float& mean, float& rstd) {
    const int tid = threadIdx.x;
    red[tid] = s; red[256 + tid] = q;
    __syncthreads();
    for (int st = 128; st > 0; st >>= 1) {
        if (tid < st) { red[tid] += red[tid + st]; red[256 + tid] += red[256 + tid + st]; }
        __syncthreads();
    }
    mean = red[0] * (1.0f / D_MODEL);
    float var = red[256] * (1.0f / D_MODEL) - mean * mean;
    rstd = rsqrtf(var + 1e-5f);
}

// -------------------------- im2col (patches -> bf16) -----------------------
__global__ __launch_bounds__(256) void im2col_kernel(const float* __restrict__ x,
                                                     unsigned short* __restrict__ out) {
    int idx = blockIdx.x * 256 + threadIdx.x;          // ROWS*D exactly
    int row = idx / D_MODEL, col = idx - row * D_MODEL;
    int b  = row / N_TOK,    n   = row - b * N_TOK;
    int gy = n / GW_GRID,    gx  = n - gy * GW_GRID;
    int c  = col >> 8,       r   = col & 255;          // c*256 + py*16 + px
    int py = r >> 4,         px  = r & 15;
    int yy = gy * 16 + py,   xx  = gx * 16 + px;
    out[idx] = f2bf(x[((b * 3 + c) * 384 + yy) * 384 + xx]);
}

// -------------------------- bf16 WMMA GEMM ---------------------------------
// C[M,N] = op(A[M,K](bf16) * B(fp32) + bias)
//   BTRANS=0: B is [K,N] row-major.   BTRANS=1: B is [N,K] row-major.
// OUT_MODE: 0 = fp32 store, 1 = bf16 store, 2 = fp32 accumulate (+=).
// Block = 256 threads = 8 waves, 64x64 C tile, K-step 32.
// Double-buffered LDS; A tile staged with async global->LDS (no VGPR pass),
// B tile staged via VALU (needs fp32->bf16 conversion + transpose).
template<bool BTRANS, bool BIAS, bool RELU, int OUT_MODE>
__global__ __launch_bounds__(256) void gemm_bf16_kernel(
        const unsigned short* __restrict__ A,
        const float* __restrict__ Bm,
        const float* __restrict__ bias,
        float* __restrict__ Cf,
        unsigned short* __restrict__ Cbf,
        int M, int N, int K) {
    __shared__ __align__(16) unsigned short As[2][64 * 32];   // [m][k]
    __shared__ __align__(16) unsigned short Bs[2][64 * 32];   // [n][k] (transposed)

    const int tid   = threadIdx.x;
    const int lane  = tid & 31;
    const int wave  = tid >> 5;
    const int m_sub = wave & 3;          // 4 M-subtiles of 16
    const int npair = wave >> 2;         // 2 groups of 32 N-columns
    const int blockM = blockIdx.x * 64;
    const int blockN = blockIdx.y * 64;

    const int arow = tid >> 2;           // 0..63  staging row
    const int ac0  = (tid & 3) * 8;      // 0,8,16,24 staging k-offset

    v8f acc[2] = { {0,0,0,0,0,0,0,0}, {0,0,0,0,0,0,0,0} };

    const int half = lane >> 4;          // 0: lanes 0-15, 1: lanes 16-31
    const int l15  = lane & 15;

    // ---- prologue: stage tile 0 into buffer 0 ----
    async_copy_b128(&A[(blockM + arow) * K + ac0], &As[0][arow * 32 + ac0]);
    {
        unsigned short tmp[8];
        if (BTRANS) {
            const float* src = &Bm[(blockN + arow) * K + ac0];
            #pragma unroll
            for (int j = 0; j < 8; ++j) tmp[j] = f2bf(src[j]);
        } else {
            #pragma unroll
            for (int j = 0; j < 8; ++j)
                tmp[j] = f2bf(Bm[(ac0 + j) * N + blockN + arow]);
        }
        *(U4*)&Bs[0][arow * 32 + ac0] = *(U4*)tmp;
    }

    int cur = 0;
    for (int k0 = 0; k0 < K; k0 += 32) {
        async_wait0();        // staged A tile (ASYNCcnt) visible in LDS
        __syncthreads();      // staged B tile (DScnt) visible to all waves

        // ---- stage next tile into the other buffer while computing ----
        if (k0 + 32 < K) {
            const int kn = k0 + 32;
            async_copy_b128(&A[(blockM + arow) * K + kn + ac0],
                            &As[cur ^ 1][arow * 32 + ac0]);
            unsigned short tmp[8];
            if (BTRANS) {
                const float* src = &Bm[(blockN + arow) * K + kn + ac0];
                #pragma unroll
                for (int j = 0; j < 8; ++j) tmp[j] = f2bf(src[j]);
                if (kn + 32 < K)   // gfx1250 global_prefetch_b8 on weight stream
                    __builtin_prefetch(&Bm[(blockN + arow) * K + kn + 32 + ac0], 0, 1);
            } else {
                #pragma unroll
                for (int j = 0; j < 8; ++j)
                    tmp[j] = f2bf(Bm[(kn + ac0 + j) * N + blockN + arow]);
                if (kn + 32 < K)
                    __builtin_prefetch(&Bm[(kn + 32 + ac0) * N + blockN + arow], 0, 1);
            }
            *(U4*)&Bs[cur ^ 1][arow * 32 + ac0] = *(U4*)tmp;
        }

        // ---- per-wave fragment loads (CDNA5 wave32 WMMA layouts) ----
        Frag fa, fb0, fb1;
        {   // A 16x32 bf16: lanes<16 rows M, K {0-7,16-23}; lanes>=16 K {8-15,24-31}
            const unsigned short* ap = &As[cur][(m_sub * 16 + l15) * 32 + half * 8];
            fa.u[0] = *(const U4*)ap;
            fa.u[1] = *(const U4*)(ap + 16);
        }
        {   // B 32x16 bf16: lane col N=l15, K {0-7,16-23} (lanes<16) / {8-15,24-31} (lanes>=16)
            const unsigned short* bp0 = &Bs[cur][(npair * 32 + l15) * 32 + half * 8];
            fb0.u[0] = *(const U4*)bp0;  fb0.u[1] = *(const U4*)(bp0 + 16);
            const unsigned short* bp1 = &Bs[cur][(npair * 32 + 16 + l15) * 32 + half * 8];
            fb1.u[0] = *(const U4*)bp1;  fb1.u[1] = *(const U4*)(bp1 + 16);
        }
        acc[0] = __builtin_amdgcn_wmma_f32_16x16x32_bf16(
                     false, fa.v, false, fb0.v, (short)0, acc[0], false, false);
        acc[1] = __builtin_amdgcn_wmma_f32_16x16x32_bf16(
                     false, fa.v, false, fb1.v, (short)0, acc[1], false, false);
        cur ^= 1;
    }

    // ---- epilogue: C layout, lane col = l15, row = r + 8*half ----
    #pragma unroll
    for (int j = 0; j < 2; ++j) {
        const int col = blockN + npair * 32 + j * 16 + l15;
        float bv = 0.0f;
        if (BIAS) bv = bias[col];
        #pragma unroll
        for (int r = 0; r < 8; ++r) {
            const int row = blockM + m_sub * 16 + r + half * 8;
            float v = acc[j][r] + bv;
            if (RELU) v = fmaxf(v, 0.0f);
            const int idx = row * N + col;
            if (OUT_MODE == 0)      Cf[idx]  = v;
            else if (OUT_MODE == 1) Cbf[idx] = f2bf(v);
            else                    Cf[idx] += v;
        }
    }
}

// -------------------------- positional embedding ---------------------------
__global__ __launch_bounds__(256) void posln_kernel(
        const float* __restrict__ pos_y, const float* __restrict__ pos_x,
        const int* __restrict__ xs_p, const int* __restrict__ ys_p,
        const float* __restrict__ w, const float* __restrict__ b,
        float* __restrict__ posln, float* __restrict__ pos_out) {
    __shared__ float red[512];
    const int n = blockIdx.x, tid = threadIdx.x;
    const int gy = n / GW_GRID, gx = n - gy * GW_GRID;
    const int ys = *ys_p, xs = *xs_p;
    float vals[3], s = 0.0f, q = 0.0f;
    #pragma unroll
    for (int e = 0; e < 3; ++e) {
        int c = tid + e * 256;
        float v = pos_y[(gy + ys) * D_MODEL + c] + pos_x[(gx + xs) * D_MODEL + c];
        vals[e] = v; s += v; q += v * v;
    }
    float mean, rstd;
    block_ln_stats(s, q, red, mean, rstd);
    #pragma unroll
    for (int e = 0; e < 3; ++e) {
        int c = tid + e * 256;
        float v = (vals[e] - mean) * rstd * w[c] + b[c];
        posln[n * D_MODEL + c] = v;
        for (int bb = 0; bb < B_BATCH; ++bb)      // broadcast: 2nd output
            pos_out[(bb * N_TOK + n) * D_MODEL + c] = v;
    }
}

// -------------------------- t0 = LN(conv_out) + pos ------------------------
__global__ __launch_bounds__(256) void embed_kernel(
        const float* __restrict__ h, const float* __restrict__ w,
        const float* __restrict__ b, const float* __restrict__ posln,
        float* __restrict__ t) {
    __shared__ float red[512];
    const int row = blockIdx.x, tid = threadIdx.x;
    const float* hp = h + row * D_MODEL;
    float vals[3], s = 0.0f, q = 0.0f;
    #pragma unroll
    for (int e = 0; e < 3; ++e) {
        int c = tid + e * 256;
        float v = hp[c];
        vals[e] = v; s += v; q += v * v;
    }
    float mean, rstd;
    block_ln_stats(s, q, red, mean, rstd);
    const int n = row % N_TOK;
    #pragma unroll
    for (int e = 0; e < 3; ++e) {
        int c = tid + e * 256;
        t[row * D_MODEL + c] = (vals[e] - mean) * rstd * w[c] + b[c]
                               + posln[n * D_MODEL + c];
    }
}

// ---- LN (optionally fused residual add t += resid), bf16 output -----------
template<bool ADD_RESID>
__global__ __launch_bounds__(256) void ln_bf16_kernel(
        float* __restrict__ t, const float* __restrict__ resid,
        const float* __restrict__ w, const float* __restrict__ b,
        unsigned short* __restrict__ out) {
    __shared__ float red[512];
    const int row = blockIdx.x, tid = threadIdx.x;
    const int base = row * D_MODEL;
    float vals[3], s = 0.0f, q = 0.0f;
    #pragma unroll
    for (int e = 0; e < 3; ++e) {
        int c = tid + e * 256;
        float v = t[base + c];
        if (ADD_RESID) { v += resid[base + c]; t[base + c] = v; }
        vals[e] = v; s += v; q += v * v;
    }
    float mean, rstd;
    block_ln_stats(s, q, red, mean, rstd);
    #pragma unroll
    for (int e = 0; e < 3; ++e) {
        int c = tid + e * 256;
        out[base + c] = f2bf((vals[e] - mean) * rstd * w[c] + b[c]);
    }
}

// -------------------------- flash attention (VALU) -------------------------
// q/k/v bf16 [ROWS, D] laid out [b*576+n, h*64+dh]. One block per (b,h,qchunk).
__global__ __launch_bounds__(256) void attention_kernel(
        const unsigned short* __restrict__ q, const unsigned short* __restrict__ k,
        const unsigned short* __restrict__ v, float* __restrict__ o) {
    __shared__ float Ks[64 * 64];
    __shared__ float Vs[64 * 64];
    const int bh = blockIdx.x;            // b*NHEAD + h
    const int b = bh / NHEAD, h = bh - b * NHEAD;
    const int tid = threadIdx.x;
    const int qrow = blockIdx.y * 256 + tid;
    const bool active = qrow < N_TOK;
    const int base = (b * N_TOK) * D_MODEL + h * DHEAD;

    float qreg[DHEAD], oacc[DHEAD];
    #pragma unroll
    for (int d = 0; d < DHEAD; ++d) oacc[d] = 0.0f;
    if (active) {
        const unsigned short* qp = &q[base + qrow * D_MODEL];
        #pragma unroll
        for (int d = 0; d < DHEAD; ++d) qreg[d] = bf2f(qp[d]);
    }
    float mmax = -1e30f, lsum = 0.0f;
    const float scale = 0.125f;           // 1/sqrt(64)

    for (int kt0 = 0; kt0 < N_TOK; kt0 += 64) {
        __syncthreads();
        #pragma unroll
        for (int i = 0; i < 16; ++i) {    // 4096 elems / 256 threads
            int idx = tid * 16 + i;
            int tok = idx >> 6, d = idx & 63;
            Ks[idx] = bf2f(k[base + (kt0 + tok) * D_MODEL + d]);
            Vs[idx] = bf2f(v[base + (kt0 + tok) * D_MODEL + d]);
        }
        __syncthreads();
        if (active) {
            for (int kk = 0; kk < 64; ++kk) {
                const float* kp = &Ks[kk * 64];
                float dot = 0.0f;
                #pragma unroll
                for (int d = 0; d < DHEAD; ++d) dot += qreg[d] * kp[d];
                float logit = dot * scale;
                float mnew = fmaxf(mmax, logit);
                float corr = __expf(mmax - mnew);
                float p    = __expf(logit - mnew);
                lsum = lsum * corr + p;
                const float* vp = &Vs[kk * 64];
                #pragma unroll
                for (int d = 0; d < DHEAD; ++d) oacc[d] = oacc[d] * corr + p * vp[d];
                mmax = mnew;
            }
        }
    }
    if (active) {
        float inv = 1.0f / lsum;
        float* op = &o[base + qrow * D_MODEL];
        #pragma unroll
        for (int d = 0; d < DHEAD; ++d) op[d] = oacc[d] * inv;
    }
}

// ---------------------------------------------------------------------------
extern "C" void kernel_launch(void* const* d_in, const int* in_sizes, int n_in,
                              void* d_out, int out_size, void* d_ws, size_t ws_size,
                              hipStream_t stream) {
    (void)in_sizes; (void)n_in; (void)out_size; (void)ws_size;
    const float* x        = (const float*)d_in[0];
    const float* conv_w   = (const float*)d_in[1];
    const float* conv_b   = (const float*)d_in[2];
    const float* ln_w     = (const float*)d_in[3];
    const float* ln_b     = (const float*)d_in[4];
    const float* pos_y    = (const float*)d_in[5];
    const float* pos_x    = (const float*)d_in[6];
    const float* mha_ln_w = (const float*)d_in[7];
    const float* mha_ln_b = (const float*)d_in[8];
    const float* qw       = (const float*)d_in[9];
    const float* kw       = (const float*)d_in[10];
    const float* vw       = (const float*)d_in[11];
    const float* ffn_ln_w = (const float*)d_in[12];
    const float* ffn_ln_b = (const float*)d_in[13];
    const float* w1       = (const float*)d_in[14];
    const float* b1       = (const float*)d_in[15];
    const float* w2       = (const float*)d_in[16];
    const float* b2       = (const float*)d_in[17];
    const int*   xs       = (const int*)d_in[18];
    const int*   ys       = (const int*)d_in[19];

    float* t_out   = (float*)d_out;                           // output 0: t
    float* pos_out = (float*)d_out + (size_t)ROWS * D_MODEL;  // output 1: pos_emb

    char* wsp = (char*)d_ws;
    auto alloc = [&](size_t bytes) {
        char* p = wsp; wsp += (bytes + 255) & ~size_t(255); return (void*)p;
    };
    unsigned short* patches = (unsigned short*)alloc((size_t)ROWS * D_MODEL * 2);
    float*          hbuf    = (float*)alloc((size_t)ROWS * D_MODEL * 4);  // conv out / attn o
    float*          posln   = (float*)alloc((size_t)N_TOK * D_MODEL * 4);
    unsigned short* flbuf   = (unsigned short*)alloc((size_t)ROWS * D_MODEL * 2);
    unsigned short* qb      = (unsigned short*)alloc((size_t)ROWS * D_MODEL * 2);
    unsigned short* kb      = (unsigned short*)alloc((size_t)ROWS * D_MODEL * 2);
    unsigned short* vb      = (unsigned short*)alloc((size_t)ROWS * D_MODEL * 2);
    unsigned short* mid     = (unsigned short*)alloc((size_t)ROWS * FFN_DIM * 2);

    const dim3 gDD(ROWS / 64, D_MODEL / 64);   // 144 x 12
    const dim3 gDF(ROWS / 64, FFN_DIM / 64);   // 144 x 48

    // 1) patch embed: im2col -> GEMM(conv_w^T, bias, relu) -> fp32 hbuf
    im2col_kernel<<<ROWS * D_MODEL / 256, 256, 0, stream>>>(x, patches);
    gemm_bf16_kernel<true, true, true, 0><<<gDD, 256, 0, stream>>>(
        patches, conv_w, conv_b, hbuf, nullptr, ROWS, D_MODEL, D_MODEL);

    // 2) positional embedding (also writes output 1) and t0 = LN(h) + pos
    posln_kernel<<<N_TOK, 256, 0, stream>>>(pos_y, pos_x, xs, ys, ln_w, ln_b,
                                            posln, pos_out);
    embed_kernel<<<ROWS, 256, 0, stream>>>(hbuf, ln_w, ln_b, posln, t_out);

    // 3) transformer layers
    for (int i = 0; i < 12; ++i) {
        const size_t wqk = (size_t)i * D_MODEL * D_MODEL;
        const size_t wf1 = (size_t)i * D_MODEL * FFN_DIM;
        // pre-LN for MHA -> bf16
        ln_bf16_kernel<false><<<ROWS, 256, 0, stream>>>(
            t_out, nullptr, mha_ln_w + i * D_MODEL, mha_ln_b + i * D_MODEL, flbuf);
        // QKV projections (bf16 out)
        gemm_bf16_kernel<false, false, false, 1><<<gDD, 256, 0, stream>>>(
            flbuf, qw + wqk, nullptr, nullptr, qb, ROWS, D_MODEL, D_MODEL);
        gemm_bf16_kernel<false, false, false, 1><<<gDD, 256, 0, stream>>>(
            flbuf, kw + wqk, nullptr, nullptr, kb, ROWS, D_MODEL, D_MODEL);
        gemm_bf16_kernel<false, false, false, 1><<<gDD, 256, 0, stream>>>(
            flbuf, vw + wqk, nullptr, nullptr, vb, ROWS, D_MODEL, D_MODEL);
        // attention -> fp32 o (reuse hbuf)
        attention_kernel<<<dim3(B_BATCH * NHEAD, (N_TOK + 255) / 256), 256, 0, stream>>>(
            qb, kb, vb, hbuf);
        // t += o, then pre-LN for FFN -> bf16
        ln_bf16_kernel<true><<<ROWS, 256, 0, stream>>>(
            t_out, hbuf, ffn_ln_w + i * D_MODEL, ffn_ln_b + i * D_MODEL, flbuf);
        // FFN1 (bias+relu, bf16 out) and FFN2 (bias, accumulate into t)
        gemm_bf16_kernel<false, true, true, 1><<<gDF, 256, 0, stream>>>(
            flbuf, w1 + wf1, b1 + (size_t)i * FFN_DIM, nullptr, mid,
            ROWS, FFN_DIM, D_MODEL);
        gemm_bf16_kernel<false, true, false, 2><<<gDD, 256, 0, stream>>>(
            mid, w2 + wf1, b2 + (size_t)i * D_MODEL, t_out, nullptr,
            ROWS, D_MODEL, FFN_DIM);
    }
}
